// EncoderRNN_80006650790166
// MI455X (gfx1250) — matrix operand order repaired and checked by reading, not compiled, not run on hardware
//
#include <hip/hip_runtime.h>
#include <hip/hip_bf16.h>

// ---------------------------------------------------------------------------
// Bidirectional LSTM encoder for MI455X (gfx1250, wave32, WMMA f16->f32).
//   VOCAB=32000, HID=512, SEQ=2048, BATCH=32
// Strategy:
//   1) gather embedding -> fp16 [S*B, 512]
//   2) pack wx/wh (f32 [512,2048]) into WMMA-B fragment layout (fp16)
//   3) big parallel GEMM: gx[dir] = emb16 @ wx[dir]  (fp16 out, [S*B, 2048])
//      4 N-tiles per wave (A fragment reuse x4)
//   4) persistent recurrent kernel: wh fragments held in 128 VGPRs per wave;
//      per step: 16 v_wmma on the critical path, grid barrier, LSTM cell,
//      grid barrier.
// ---------------------------------------------------------------------------

#define SEQ   2048
#define BATCH 32
#define HID   512
#define GSZ   (4 * HID)          // 2048 gate columns
#define NBLK  64                 // persistent blocks (x 256 thr = 512 waves)

typedef __attribute__((ext_vector_type(8)))  _Float16 v8h;
typedef __attribute__((ext_vector_type(16))) _Float16 v16h;
typedef __attribute__((ext_vector_type(8)))  float    v8f;

__device__ __forceinline__ v16h cat8(v8h lo, v8h hi) {
  return __builtin_shufflevector(lo, hi, 0,1,2,3,4,5,6,7,8,9,10,11,12,13,14,15);
}

// A fragment (16x32 f16, row-major source, lda halves):
//   lane<16 : M=lane,    halves = K[k0+0..7]   and K[k0+16..23]
//   lane>=16: M=lane-16, halves = K[k0+8..15]  and K[k0+24..31]
__device__ __forceinline__ v16h load_afrag(const _Float16* __restrict__ A, int lda,
                                           int row0, int k0, int lane) {
  int m   = lane & 15;
  int sel = lane >> 4;
  const _Float16* p = A + (size_t)(row0 + m) * lda + k0 + sel * 8;
  v8h lo = *(const v8h*)(p);
  v8h hi = *(const v8h*)(p + 16);
  return cat8(lo, hi);
}

// B fragment from pre-packed buffer: block (nt,kt) holds 32 lanes x 16 halves
// contiguously; each lane does two contiguous 16B loads.
__device__ __forceinline__ v16h load_bfrag(const _Float16* __restrict__ P,
                                           int nt, int kt, int lane) {
  const _Float16* p = P + (((size_t)nt * 16 + kt) * 512) + lane * 16;
  v8h lo = *(const v8h*)(p);
  v8h hi = *(const v8h*)(p + 8);
  return cat8(lo, hi);
}

__device__ __forceinline__ float sigf(float x) {
  return 1.0f / (1.0f + __expf(-x));
}
__device__ __forceinline__ float tanh_fast(float x) {
  float e = __expf(2.0f * x);
  return 1.0f - 2.0f / (e + 1.0f);     // stable at +/- inf
}

__device__ __forceinline__ void grid_barrier(unsigned* bar, unsigned target) {
  __threadfence();
  __syncthreads();
  if (threadIdx.x == 0) {
    atomicAdd(bar, 1u);
    while (*(volatile unsigned*)bar < target) {
      __builtin_amdgcn_s_sleep(1);
    }
  }
  __syncthreads();
  __threadfence();
}

// ---------------------------------------------------------------------------
// 1) embedding gather (int64 tokens) -> fp16 rows, 8 halves per thread
// ---------------------------------------------------------------------------
__global__ void __launch_bounds__(256) embed_gather_kernel(
    const long long* __restrict__ tokens, const float* __restrict__ emb,
    _Float16* __restrict__ emb16) {
  size_t idx = (size_t)blockIdx.x * 256 + threadIdx.x;   // S*B*64 threads
  size_t row = idx >> 6;                                 // token row (s*32+b)
  int    hb  = (int)(idx & 63) * 8;
  long long tok = tokens[row];
  const float* src = emb + (size_t)tok * HID + hb;
  _Float16*    dst = emb16 + row * HID + hb;
#pragma unroll
  for (int j = 0; j < 8; ++j) dst[j] = (_Float16)src[j];
}

// ---------------------------------------------------------------------------
// 2) pack f32 [512, 2048] weight into WMMA-B fragment layout (fp16)
//    block (nt,kt): lane L, half j -> B[k][n], n = nt*16 + (L&15),
//    k = kt*32 + (j<8 ? 8*(L>>4)+j : 16+8*(L>>4)+(j-8))
// ---------------------------------------------------------------------------
__global__ void __launch_bounds__(256) pack_weights_kernel(
    const float* __restrict__ w0, const float* __restrict__ w1,
    const float* __restrict__ w2, const float* __restrict__ w3,
    _Float16* __restrict__ p0, _Float16* __restrict__ p1,
    _Float16* __restrict__ p2, _Float16* __restrict__ p3) {
  const float* w; _Float16* p;
  switch (blockIdx.y) {
    case 0:  w = w0; p = p0; break;
    case 1:  w = w1; p = p1; break;
    case 2:  w = w2; p = p2; break;
    default: w = w3; p = p3; break;
  }
  int idx  = blockIdx.x * 256 + threadIdx.x;   // 512*2048 elements
  int j    = idx & 15;
  int lane = (idx >> 4) & 31;
  int kt   = (idx >> 9) & 15;
  int nt   = idx >> 13;                        // 0..127
  int n = nt * 16 + (lane & 15);
  int k = kt * 32 + ((j < 8) ? 8 * (lane >> 4) + j : 16 + 8 * (lane >> 4) + (j - 8));
  p[(((size_t)nt * 16 + kt) * 512) + lane * 16 + j] = (_Float16)w[(size_t)k * GSZ + n];
}

// ---------------------------------------------------------------------------
// 3) big parallel GEMM: gx[dir] = emb16[S*B,512] @ wx_packed -> fp16 [S*B,2048]
//    each wave: one 16-row M tile x four 16-col N tiles; A fragment reused 4x
// ---------------------------------------------------------------------------
__global__ void __launch_bounds__(256) gemm_gx_kernel(
    const _Float16* __restrict__ emb16,
    const _Float16* __restrict__ pxf, const _Float16* __restrict__ pxb,
    _Float16* __restrict__ gxf, _Float16* __restrict__ gxb) {
  int lane = threadIdx.x & 31;
  int wave = (blockIdx.x * 256 + threadIdx.x) >> 5;  // 0 .. 4096*32-1
  int mt = wave >> 5;                                // 0..4095 (rows of 16)
  int ng = wave & 31;                                // N-group: tiles 4*ng..4*ng+3
  const _Float16* W  = blockIdx.y ? pxb : pxf;
  _Float16*       gx = blockIdx.y ? gxb : gxf;

  v8f acc[4] = {};
#pragma unroll
  for (int kt = 0; kt < 16; ++kt) {
    v16h a = load_afrag(emb16, HID, mt * 16, kt * 32, lane);
#pragma unroll
    for (int i = 0; i < 4; ++i) {
      v16h b = load_bfrag(W, ng * 4 + i, kt, lane);
      acc[i] = __builtin_amdgcn_wmma_f32_16x16x32_f16(false, a, false, b,
                                                      (short)0, acc[i], false, false);
    }
  }
  int N = lane & 15, sel = lane >> 4;
#pragma unroll
  for (int i = 0; i < 4; ++i) {
    size_t col = (size_t)(ng * 4 + i) * 16 + N;
#pragma unroll
    for (int r = 0; r < 8; ++r) {
      int M = r + 8 * sel;
      gx[(size_t)(mt * 16 + M) * GSZ + col] = (_Float16)acc[i][r];
    }
  }
}

// ---------------------------------------------------------------------------
// 4a) init: zero the grid-barrier counter, seed h/c state from h0
// ---------------------------------------------------------------------------
__global__ void __launch_bounds__(256) init_state_kernel(
    const float* __restrict__ h0, _Float16* __restrict__ h16,
    float* __restrict__ cbuf, unsigned* __restrict__ bar) {
  int i = blockIdx.x * 256 + threadIdx.x;     // 0 .. 16383 (= 32*512)
  if (i == 0) *bar = 0u;
  float v = h0[i];
  h16[i]          = (_Float16)v;   // forward h
  h16[16384 + i]  = (_Float16)v;   // backward h
  cbuf[i]         = v;             // forward c  (reference carry = (h0,h0))
  cbuf[16384 + i] = v;             // backward c
}

// ---------------------------------------------------------------------------
// 4b) persistent recurrent kernel: both directions, 2048 steps.
//     wh fragments for this wave's tile are loaded ONCE into 128 VGPRs and
//     stay register-resident across all timesteps -> per-step critical path
//     is 32 small A-loads + 16 back-to-back WMMAs.
// ---------------------------------------------------------------------------
__global__ void __launch_bounds__(256, 1) lstm_recurrent_kernel(
    const _Float16* __restrict__ gxf, const _Float16* __restrict__ gxb,
    const _Float16* __restrict__ whF, const _Float16* __restrict__ whB,
    const float* __restrict__ bxf, const float* __restrict__ bhf,
    const float* __restrict__ bxb, const float* __restrict__ bhb,
    _Float16* __restrict__ h16, float* __restrict__ cbuf,
    float* __restrict__ gates, unsigned* __restrict__ bar,
    float* __restrict__ out) {
  int lane = threadIdx.x & 31;
  int tid  = blockIdx.x * 256 + threadIdx.x;      // 0..16383
  int wave = tid >> 5;                            // 0..511
  int dir  = wave >> 8;                           // 0 fw, 1 bw
  int tile = wave & 255;
  int mt   = tile >> 7;                           // 0..1  (batch rows 0-15/16-31)
  int nt   = tile & 127;                          // 0..127 (gate cols)

  const _Float16* W  = dir ? whB : whF;
  const float*    bx = dir ? bxb : bxf;
  const float*    bh = dir ? bhb : bhf;
  const _Float16* gx = dir ? gxb : gxf;
  const _Float16* hprev = h16 + (size_t)dir * BATCH * HID;
  float*          gt    = gates + (size_t)dir * BATCH * GSZ;

  int N = lane & 15, sel = lane >> 4;
  int col = nt * 16 + N;
  float bias = bx[col] + bh[col];

  // ---- load this wave's 16 wh B-fragments once; keep register-resident ----
  v16h bfr[16];
#pragma unroll
  for (int kt = 0; kt < 16; ++kt) bfr[kt] = load_bfrag(W, nt, kt, lane);

  unsigned target = 0;
  for (int t = 0; t < SEQ; ++t) {
    int s = dir ? (SEQ - 1 - t) : t;

    // prefetch next step's gx rows (speculative, GL2)
    if (t + 1 < SEQ) {
      int sn = dir ? (SEQ - 2 - t) : (t + 1);
      __builtin_prefetch(&gx[((size_t)sn * BATCH + mt * 16) * GSZ + col], 0, 1);
    }

    // ---- recurrent GEMM: h_prev[32,512] @ wh[512,2048], one 16x16 tile ----
    v8f acc = {};
#pragma unroll
    for (int kt = 0; kt < 16; ++kt) {
      v16h a = load_afrag(hprev, HID, mt * 16, kt * 32, lane);
      acc = __builtin_amdgcn_wmma_f32_16x16x32_f16(false, a, false, bfr[kt],
                                                   (short)0, acc, false, false);
    }
#pragma unroll
    for (int r = 0; r < 8; ++r) {
      int brow = mt * 16 + r + 8 * sel;          // batch index 0..31
      float g = acc[r] + bias + (float)gx[((size_t)s * BATCH + brow) * GSZ + col];
      gt[(size_t)brow * GSZ + col] = g;
    }

    target += NBLK;
    grid_barrier(bar, target);

    // ---- elementwise LSTM cell: 2 elements per thread (2*32*512 total) ----
#pragma unroll
    for (int j = 0; j < 2; ++j) {
      int e   = tid + j * 16384;                 // 0..32767
      int d2  = e >> 14;
      int rem = e & 16383;
      int b   = rem >> 9;
      int hc  = rem & 511;
      const float* gE = gates + (size_t)d2 * BATCH * GSZ + (size_t)b * GSZ;
      float iv = sigf(gE[hc]);
      float fv = sigf(gE[HID + hc]);
      float gv = tanh_fast(gE[2 * HID + hc]);
      float ov = sigf(gE[3 * HID + hc]);
      float* cE = cbuf + (size_t)d2 * BATCH * HID + (size_t)b * HID + hc;
      float c2 = fv * (*cE) + iv * gv;
      float h2 = ov * tanh_fast(c2);
      *cE = c2;
      int s2 = d2 ? (SEQ - 1 - t) : t;
      out[(size_t)b * (SEQ * 2 * HID) + (size_t)s2 * (2 * HID) + d2 * HID + hc] = h2;
      h16[(size_t)d2 * BATCH * HID + (size_t)b * HID + hc] = (_Float16)h2;
    }

    target += NBLK;
    grid_barrier(bar, target);
  }
}

// ---------------------------------------------------------------------------
extern "C" void kernel_launch(void* const* d_in, const int* in_sizes, int n_in,
                              void* d_out, int out_size, void* d_ws, size_t ws_size,
                              hipStream_t stream) {
  const long long* tokens    = (const long long*)d_in[0];
  const float*     h0        = (const float*)d_in[1];
  const float*     embedding = (const float*)d_in[2];
  const float*     wx_f      = (const float*)d_in[3];
  const float*     bx_f      = (const float*)d_in[4];
  const float*     wh_f      = (const float*)d_in[5];
  const float*     bh_f      = (const float*)d_in[6];
  const float*     wx_b      = (const float*)d_in[7];
  const float*     bx_b      = (const float*)d_in[8];
  const float*     wh_b      = (const float*)d_in[9];
  const float*     bh_b      = (const float*)d_in[10];
  float*           out       = (float*)d_out;

  // ---- workspace carve-up (256B aligned) ----
  char*  ws  = (char*)d_ws;
  size_t off = 0;
  auto carve = [&](size_t bytes) -> char* {
    char* p = ws + off;
    off = (off + bytes + 255) & ~(size_t)255;
    return p;
  };
  _Float16* emb16 = (_Float16*)carve((size_t)SEQ * BATCH * HID * 2);     //  64 MB
  _Float16* gxf   = (_Float16*)carve((size_t)SEQ * BATCH * GSZ * 2);     // 256 MB
  _Float16* gxb   = (_Float16*)carve((size_t)SEQ * BATCH * GSZ * 2);     // 256 MB
  _Float16* pxf   = (_Float16*)carve((size_t)HID * GSZ * 2);             //   2 MB
  _Float16* phf   = (_Float16*)carve((size_t)HID * GSZ * 2);
  _Float16* pxb   = (_Float16*)carve((size_t)HID * GSZ * 2);
  _Float16* phb   = (_Float16*)carve((size_t)HID * GSZ * 2);
  _Float16* h16   = (_Float16*)carve((size_t)2 * BATCH * HID * 2);       //  64 KB
  float*    cbuf  = (float*)   carve((size_t)2 * BATCH * HID * 4);       // 128 KB
  float*    gates = (float*)   carve((size_t)2 * BATCH * GSZ * 4);       // 512 KB
  unsigned* bar   = (unsigned*)carve(256);

  // 1) gather embeddings -> fp16
  embed_gather_kernel<<<(SEQ * BATCH * HID / 8) / 256, 256, 0, stream>>>(
      tokens, embedding, emb16);

  // 2) pack all four weight matrices into WMMA-B fragment layout
  pack_weights_kernel<<<dim3((HID * GSZ) / 256, 4), 256, 0, stream>>>(
      wx_f, wh_f, wx_b, wh_b, pxf, phf, pxb, phb);

  // 3) batched input-projection GEMM (both directions)
  //    4096 M-tiles x 32 N-groups, 8 waves/block -> 16384 blocks per dir
  gemm_gx_kernel<<<dim3(16384, 2), 256, 0, stream>>>(emb16, pxf, pxb, gxf, gxb);

  // 4) recurrent phase
  init_state_kernel<<<64, 256, 0, stream>>>(h0, h16, cbuf, bar);
  lstm_recurrent_kernel<<<NBLK, 256, 0, stream>>>(
      gxf, gxb, phf, phb, bx_f, bh_f, bx_b, bh_b,
      h16, cbuf, gates, bar, out);
}
